// GTEncoderLayer_31052613550224
// MI455X (gfx1250) — matrix operand order repaired and checked by reading, not compiled, not run on hardware
//
#include <hip/hip_runtime.h>
#include <hip/hip_bf16.h>

#define BB 4
#define NS 512
#define DD 64
#define HH 8
#define DKK 8
#define DHID 128
#define LNEPS 1e-5f

typedef __attribute__((ext_vector_type(16))) _Float16 v16h;
typedef __attribute__((ext_vector_type(8)))  float    v8f;

// ---- CDNA5 async global->LDS (guarded: falls back to plain loads) -----------
#if defined(__gfx1250__) && __has_builtin(__builtin_amdgcn_global_load_async_to_lds_b128) && __has_builtin(__builtin_amdgcn_s_wait_asynccnt)
#define USE_ASYNC_LDS 1
typedef __attribute__((__vector_size__(16))) int v4i_t;
typedef __attribute__((address_space(1))) v4i_t gv4i_t;
typedef __attribute__((address_space(3))) v4i_t lv4i_t;
static __device__ __forceinline__ void async_b128(const float* g, float* l) {
  __builtin_amdgcn_global_load_async_to_lds_b128((gv4i_t*)g, (lv4i_t*)l, 0, 0);
}
static __device__ __forceinline__ void async_wait() {
  __builtin_amdgcn_s_wait_asynccnt(0);
}
#else
#define USE_ASYNC_LDS 0
#endif

// ---- WMMA helpers (CDNA5 16x16x32 f16 -> f32) -------------------------------
// A (16x32 f16): lane l: row m=l&15, half=l>>4; v16h elem 2v,2v+1 <->
//   K = 16*half + 2*v (+1)
static __device__ __forceinline__ v16h load_a_frag(const _Float16* tile, int ld,
                                                   int lane, int kb) {
  int m = lane & 15, hl = lane >> 4;
  v16h a;
#pragma unroll
  for (int v = 0; v < 8; ++v) {
    int kk = 16 * hl + 2 * v;
    a[2 * v]     = tile[m * ld + kb + kk];
    a[2 * v + 1] = tile[m * ld + kb + kk + 1];
  }
  return a;
}

// B fragments stored fragment-major in LDS: [frag][lane][16 halfs] contiguous,
// so each lane loads 32 contiguous bytes (-> 2x ds_load_b128).
static __device__ __forceinline__ v16h load_b_packed(const _Float16* base,
                                                     int frag, int lane) {
  return *(const v16h*)(base + ((frag * 32 + lane) << 4));
}

static __device__ __forceinline__ v8f wmma_f16(v16h a, v16h b, v8f c) {
  return __builtin_amdgcn_wmma_f32_16x16x32_f16(false, a, false, b, (short)0, c,
                                                false, false);
}

// Compiler-level barrier for intra-wave LDS communication (LDS in-order per wave).
static __device__ __forceinline__ void wave_sync() {
  __asm__ volatile("" ::: "memory");
  __builtin_amdgcn_wave_barrier();
  __asm__ volatile("" ::: "memory");
}

// ---- Kernel 1: node LN + QKV ------------------------------------------------
__global__ __launch_bounds__(64) void k_node_qkv(
    const float* __restrict__ nin, const float* __restrict__ g1,
    const float* __restrict__ b1, const float* __restrict__ wqkv,
    float* __restrict__ Qw, float* __restrict__ Kw, float* __restrict__ Vw) {
  int t = blockIdx.x;            // b*N + i
  int b = t >> 9, i = t & 511;
  int c = threadIdx.x;
  __shared__ float sx[DD];
  __shared__ float sxn[DD];
  __shared__ float smv[2];
  float xv = nin[(size_t)t * DD + c];
  sx[c] = xv;
  __syncthreads();
  if (c == 0) {
    float m = 0.f;
    for (int k = 0; k < DD; ++k) m += sx[k];
    m *= (1.f / DD);
    float v = 0.f;
    for (int k = 0; k < DD; ++k) { float d = sx[k] - m; v += d * d; }
    v *= (1.f / DD);
    smv[0] = m; smv[1] = rsqrtf(v + LNEPS);
  }
  __syncthreads();
  sxn[c] = (xv - smv[0]) * smv[1] * g1[c] + b1[c];
  __syncthreads();
#pragma unroll
  for (int s = 0; s < 3; ++s) {
    int o = s * 64 + c;
    float acc = 0.f;
    for (int k = 0; k < DD; ++k) acc += sxn[k] * wqkv[o * DD + k];
    int h = c >> 3, dki = c & 7;
    float* dst = (s == 0) ? Qw : (s == 1 ? Kw : Vw);
    dst[(((size_t)(b * HH + h) * NS) + i) * DKK + dki] = acc;
  }
}

// ---- Kernel 2: edge LN + E/G projection (WMMA) + dyn_cent -------------------
__global__ __launch_bounds__(256) void k_edge_proj(
    const float* __restrict__ e, const float* __restrict__ g1,
    const float* __restrict__ b1, const float* __restrict__ we,
    const float* __restrict__ wg, float* __restrict__ Ews,
    float* __restrict__ dyn) {
  int t = blockIdx.x;            // b*N + i  (one full edge row of 512 j's)
  int b = t >> 9, i = t & 511;
  int tid = threadIdx.x, lane = tid & 31, wave = tid >> 5;
  int m = lane & 15, hl = lane >> 4;

  // combined weight [64 x 16] (cols 0..7 = w_e, 8..15 = w_g), fragment-major
  __shared__ __attribute__((aligned(32))) _Float16 sWeg[2 * 32 * 16];
  __shared__ float sG[DD], sB[DD];
  __shared__ __attribute__((aligned(16))) float sRaw[8][16 * DD];   // 32 KB
  __shared__ _Float16 sXe[8][16 * DD];   // 16 KB
  __shared__ float sC[8][16 * 16];       // 8 KB
  __shared__ float sGs[8][HH];

  for (int idx = tid; idx < 2 * 32 * 16; idx += 256) {
    int kk = idx & 15, ln = (idx >> 4) & 31, kc = idx >> 9;
    int k = kc * 32 + 16 * (ln >> 4) + kk;
    int n = ln & 15;
    float w = (n < 8) ? we[n * DD + k] : wg[(n - 8) * DD + k];
    sWeg[idx] = (_Float16)w;
  }
  if (tid < DD) { sG[tid] = g1[tid]; sB[tid] = b1[tid]; }
  __syncthreads();

  v16h bw0 = load_b_packed(sWeg, 0, lane);
  v16h bw1 = load_b_packed(sWeg, 1, lane);

  const size_t rowbase = (size_t)(b * NS + i) * NS;   // token index of (b,i,0)
  float pg = 0.f;

  for (int tile = 0; tile < 4; ++tile) {
    int jbase = (wave * 4 + tile) * 16;
    const float* ept = e + (rowbase + jbase) * DD;
#if USE_ASYNC_LDS
#pragma unroll
    for (int s = 0; s < 8; ++s)
      async_b128(ept + s * 128 + lane * 4, &sRaw[wave][s * 128 + lane * 4]);
    async_wait();
    wave_sync();
#else
#pragma unroll 4
    for (int s = 0; s < 32; ++s) sRaw[wave][s * 32 + lane] = ept[s * 32 + lane];
    wave_sync();
#endif
    // LN stats for token m (redundant across halves)
    float mu = 0.f;
    for (int k = 0; k < DD; ++k) mu += sRaw[wave][m * DD + k];
    mu *= (1.f / DD);
    float var = 0.f;
    for (int k = 0; k < DD; ++k) { float d = sRaw[wave][m * DD + k] - mu; var += d * d; }
    var *= (1.f / DD);
    float rstd = rsqrtf(var + LNEPS);
    for (int kk = 0; kk < 32; ++kk) {
      int k = hl * 32 + kk;
      sXe[wave][m * DD + k] =
          (_Float16)((sRaw[wave][m * DD + k] - mu) * rstd * sG[k] + sB[k]);
    }
    wave_sync();
    v16h a0 = load_a_frag(&sXe[wave][0], DD, lane, 0);
    v16h a1 = load_a_frag(&sXe[wave][0], DD, lane, 32);
    v8f acc = {0.f, 0.f, 0.f, 0.f, 0.f, 0.f, 0.f, 0.f};
    acc = wmma_f16(a0, bw0, acc);
    acc = wmma_f16(a1, bw1, acc);
    // stage C tile: row = token, col = n
#pragma unroll
    for (int r = 0; r < 8; ++r) sC[wave][(r + 8 * hl) * 16 + m] = acc[r];
    if (m >= 8) {  // gate columns -> partial sigmoid sum (fast rcp)
#pragma unroll
      for (int r = 0; r < 8; ++r)
        pg += __builtin_amdgcn_rcpf(1.f + __expf(-acc[r]));
    }
    wave_sync();
    // coalesced E writes: E[b,h,i,jbase..jbase+15]
#pragma unroll
    for (int s = 0; s < 4; ++s) {
      int idx = s * 32 + lane;     // 0..127
      int h = idx >> 4, jm = idx & 15;
      Ews[((size_t)(b * HH + h) * NS + i) * NS + jbase + jm] = sC[wave][jm * 16 + h];
    }
    wave_sync();
  }
  // combine halves: lane l and l^16 share same column n
  pg += __shfl_xor(pg, 16);
  if (hl == 0 && m >= 8) sGs[wave][m - 8] = pg;
  __syncthreads();
  if (tid < HH) {
    float s = 0.f;
    for (int w = 0; w < 8; ++w) s += sGs[w][tid];
    dyn[(size_t)(b * HH + tid) * NS + i] = log1pf(s);
  }
}

// ---- Kernel 3: attention row: logits (in-place into Ews), softmax, A@V ------
__global__ __launch_bounds__(256) void k_attn(
    const float* __restrict__ Qw, const float* __restrict__ Kw,
    const float* __restrict__ Vw, float* __restrict__ Ews,
    const float* __restrict__ dyn, float* __restrict__ Vc) {
  int blk = blockIdx.x;          // (b*H+h)*N + i
  int bh = blk >> 9, i = blk & 511;
  int b = bh >> 3, h = bh & 7;
  int tid = threadIdx.x;
  __shared__ float sq[DKK];
  __shared__ float sred[256];
  __shared__ float sacc[256][8];
  if (tid < DKK)
    sq[tid] = Qw[((size_t)bh * NS + i) * DKK + tid] * 0.35355339059327373f; // dk^-0.5
  __syncthreads();
  float* Erow = Ews + (size_t)bh * NS * NS + (size_t)i * NS;
  float ev[2];
  float lmax = -1e30f;
#pragma unroll
  for (int s = 0; s < 2; ++s) {
    int j = tid + s * 256;
    const float* kp = Kw + ((size_t)bh * NS + j) * DKK;
    float d = 0.f;
#pragma unroll
    for (int k = 0; k < DKK; ++k) d += sq[k] * kp[k];
    d = fminf(fmaxf(d, -5.f), 5.f);
    float le = d + Erow[j];
    Erow[j] = le;                // store _E back for the edge-output pass
    ev[s] = le;
    lmax = fmaxf(lmax, le);
  }
  sred[tid] = lmax; __syncthreads();
  for (int st = 128; st > 0; st >>= 1) {
    if (tid < st) sred[tid] = fmaxf(sred[tid], sred[tid + st]);
    __syncthreads();
  }
  float mx = sred[0]; __syncthreads();
  float p[2]; float lsum = 0.f;
#pragma unroll
  for (int s = 0; s < 2; ++s) { p[s] = __expf(ev[s] - mx); lsum += p[s]; }
  sred[tid] = lsum; __syncthreads();
  for (int st = 128; st > 0; st >>= 1) {
    if (tid < st) sred[tid] += sred[tid + st];
    __syncthreads();
  }
  float inv = __builtin_amdgcn_rcpf(sred[0]);
  float acc[8];
#pragma unroll
  for (int k = 0; k < 8; ++k) acc[k] = 0.f;
#pragma unroll
  for (int s = 0; s < 2; ++s) {
    int j = tid + s * 256;
    const float* vp = Vw + ((size_t)bh * NS + j) * DKK;
    float a = p[s] * inv;
#pragma unroll
    for (int k = 0; k < 8; ++k) acc[k] += a * vp[k];
  }
#pragma unroll
  for (int k = 0; k < 8; ++k) sacc[tid][k] = acc[k];
  __syncthreads();
  for (int st = 128; st > 0; st >>= 1) {
    if (tid < st)
#pragma unroll
      for (int k = 0; k < 8; ++k) sacc[tid][k] += sacc[tid + st][k];
    __syncthreads();
  }
  if (tid < 8) {
    float dc = dyn[(size_t)bh * NS + i];
    Vc[((size_t)(b * NS + i)) * DD + h * DKK + tid] = sacc[0][tid] * dc;
  }
}

// ---- Kernel 4: node output: residual + o_n proj + LN + MLP ------------------
__global__ __launch_bounds__(64) void k_node_out(
    const float* __restrict__ nin, const float* __restrict__ Vc,
    const float* __restrict__ on, const float* __restrict__ g2,
    const float* __restrict__ b2, const float* __restrict__ w1,
    const float* __restrict__ w2, float* __restrict__ outN) {
  int t = blockIdx.x;
  int c = threadIdx.x;
  __shared__ float sv[DD], sx[DD], sh[DHID], smv[2];
  sv[c] = Vc[(size_t)t * DD + c];
  __syncthreads();
  float a = 0.f;
  for (int k = 0; k < DD; ++k) a += sv[k] * on[c * DD + k];
  float n1 = nin[(size_t)t * DD + c] + a;
  sx[c] = n1;
  __syncthreads();
  if (c == 0) {
    float m = 0.f;
    for (int k = 0; k < DD; ++k) m += sx[k];
    m *= (1.f / DD);
    float v = 0.f;
    for (int k = 0; k < DD; ++k) { float d = sx[k] - m; v += d * d; }
    v *= (1.f / DD);
    smv[0] = m; smv[1] = rsqrtf(v + LNEPS);
  }
  __syncthreads();
  float xn = (n1 - smv[0]) * smv[1] * g2[c] + b2[c];
  __syncthreads();
  sx[c] = xn;
  __syncthreads();
  for (int o = c; o < DHID; o += DD) {
    float acc = 0.f;
    for (int k = 0; k < DD; ++k) acc += sx[k] * w1[o * DD + k];
    sh[o] = fmaxf(acc, 0.f);
  }
  __syncthreads();
  float acc2 = 0.f;
  for (int k = 0; k < DHID; ++k) acc2 += sh[k] * w2[c * DHID + k];
  outN[(size_t)t * DD + c] = n1 + fmaxf(acc2, 0.f);
}

// ---- Kernel 5: edge output: residual(o_e @ _E) + LN + MLP (WMMA) ------------
#define EWAVES 4
__global__ __launch_bounds__(EWAVES * 32) void k_edge_out(
    const float* __restrict__ e, const float* __restrict__ Ews,
    const float* __restrict__ oe, const float* __restrict__ g2,
    const float* __restrict__ b2, const float* __restrict__ w1,
    const float* __restrict__ w2, float* __restrict__ outE) {
  int blk = blockIdx.x;
  int jt = blk & 7;              // 8 j-tiles of 64
  int bi = blk >> 3;
  int b = bi >> 9, i = bi & 511;
  int jbase = jt * 64;
  int tid = threadIdx.x, lane = tid & 31, wave = tid >> 5;
  int m = lane & 15, hl = lane >> 4;
  const int NT = EWAVES * 32;

  // W1: 16 fragments (f = nn*2+kc), W2: 16 fragments (f = nn*4+kc), frag-major
  __shared__ __attribute__((aligned(32))) _Float16 sW1[DD * DHID];   // 16 KB
  __shared__ __attribute__((aligned(32))) _Float16 sW2[DHID * DD];   // 16 KB
  __shared__ float sOE[DD * HH];             // 2 KB
  __shared__ float sG[DD], sB[DD];
  __shared__ float sEh[HH][64];              // 2 KB (_E rows for this j-range)
  __shared__ _Float16 sE1[EWAVES][16 * DD];  // 8 KB  (e + e_attn)
  __shared__ __attribute__((aligned(16))) _Float16 sXH[EWAVES][16 * DHID]; // 16 KB

  for (int idx = tid; idx < DD * DHID; idx += NT) {
    int kk = idx & 15, ln = (idx >> 4) & 31, f = idx >> 9;
    int kc = f & 1, nc = f >> 1;
    int k = kc * 32 + 16 * (ln >> 4) + kk;
    int o = nc * 16 + (ln & 15);
    sW1[idx] = (_Float16)w1[o * DD + k];
  }
  for (int idx = tid; idx < DHID * DD; idx += NT) {
    int kk = idx & 15, ln = (idx >> 4) & 31, f = idx >> 9;
    int kc = f & 3, nc = f >> 2;
    int k = kc * 32 + 16 * (ln >> 4) + kk;
    int o = nc * 16 + (ln & 15);
    sW2[idx] = (_Float16)w2[o * DHID + k];
  }
  for (int idx = tid; idx < DD * HH; idx += NT) sOE[idx] = oe[idx];
  if (tid < DD) { sG[tid] = g2[tid]; sB[tid] = b2[tid]; }
  for (int idx = tid; idx < HH * 64; idx += NT) {
    int h = idx >> 6, jl = idx & 63;
    sEh[h][jl] = Ews[((size_t)(b * HH + h) * NS + i) * NS + jbase + jl];
  }
  __syncthreads();

  size_t gtok = ((size_t)(b * NS + i)) * NS + jbase + wave * 16;
  const float* ept = e + gtok * DD;
  int jl0 = wave * 16;

  // Step 0: stage raw e tile into LDS, overlaying sXH (raw dies before X lives)
  float* rawp = (float*)&sXH[wave][0];       // 16*64 f32 == 16*128 f16 bytes
#if USE_ASYNC_LDS
#pragma unroll
  for (int s = 0; s < 8; ++s)
    async_b128(ept + s * 128 + lane * 4, rawp + s * 128 + lane * 4);
  async_wait();
  wave_sync();
#else
  __builtin_prefetch(ept + 16 * DD, 0, 1);
#pragma unroll 4
  for (int s = 0; s < 32; ++s) rawp[s * 32 + lane] = ept[s * 32 + lane];
  wave_sync();
#endif

  // Step 1: e1 = e + e_attn,  e_attn[c] = sum_h o_e[c][h] * _E[b,h,i,j]
#pragma unroll 4
  for (int s = 0; s < 32; ++s) {
    int off = s * 32 + lane;
    int mm = off >> 6, cc = off & 63;
    float ev = rawp[off];
    float att = 0.f;
#pragma unroll
    for (int h = 0; h < HH; ++h) att += sOE[cc * HH + h] * sEh[h][jl0 + mm];
    sE1[wave][off] = (_Float16)(ev + att);
  }
  wave_sync();

  // Step 2: LN -> X (f16) into sXH[0 .. 16*64) (overwrites dead raw data)
  float mu = 0.f;
  for (int k = 0; k < DD; ++k) mu += (float)sE1[wave][m * DD + k];
  mu *= (1.f / DD);
  float var = 0.f;
  for (int k = 0; k < DD; ++k) {
    float d = (float)sE1[wave][m * DD + k] - mu;
    var += d * d;
  }
  var *= (1.f / DD);
  float rstd = rsqrtf(var + LNEPS);
  for (int kk = 0; kk < 32; ++kk) {
    int k = hl * 32 + kk;
    sXH[wave][m * DD + k] =
        (_Float16)(((float)sE1[wave][m * DD + k] - mu) * rstd * sG[k] + sB[k]);
  }
  wave_sync();

  v16h a0 = load_a_frag(&sXH[wave][0], DD, lane, 0);
  v16h a1 = load_a_frag(&sXH[wave][0], DD, lane, 32);
  wave_sync();   // A in registers; sXH will be overwritten by H below

  // GEMM1: H = relu(X @ W1^T), 16x128
#pragma unroll
  for (int nn = 0; nn < 8; ++nn) {
    v8f acc = {0.f, 0.f, 0.f, 0.f, 0.f, 0.f, 0.f, 0.f};
    acc = wmma_f16(a0, load_b_packed(sW1, nn * 2 + 0, lane), acc);
    acc = wmma_f16(a1, load_b_packed(sW1, nn * 2 + 1, lane), acc);
#pragma unroll
    for (int r = 0; r < 8; ++r)
      sXH[wave][(r + 8 * hl) * DHID + nn * 16 + m] = (_Float16)fmaxf(acc[r], 0.f);
  }
  wave_sync();

  // GEMM2: Y = relu(H @ W2^T) + e1, 16x64 -> global
  v16h ha[4];
#pragma unroll
  for (int kt = 0; kt < 4; ++kt) ha[kt] = load_a_frag(&sXH[wave][0], DHID, lane, kt * 32);
#pragma unroll
  for (int nn = 0; nn < 4; ++nn) {
    v8f acc = {0.f, 0.f, 0.f, 0.f, 0.f, 0.f, 0.f, 0.f};
#pragma unroll
    for (int kt = 0; kt < 4; ++kt)
      acc = wmma_f16(ha[kt], load_b_packed(sW2, nn * 4 + kt, lane), acc);
#pragma unroll
    for (int r = 0; r < 8; ++r) {
      int mm = r + 8 * hl, cc = nn * 16 + m;
      float y = fmaxf(acc[r], 0.f) + (float)sE1[wave][mm * DD + cc];
      outE[(gtok + mm) * DD + cc] = y;
    }
  }
}

// ---- Launch -----------------------------------------------------------------
extern "C" void kernel_launch(void* const* d_in, const int* in_sizes, int n_in,
                              void* d_out, int out_size, void* d_ws, size_t ws_size,
                              hipStream_t stream) {
  (void)in_sizes; (void)n_in; (void)out_size; (void)ws_size;
  const float* n_in_p = (const float*)d_in[0];
  const float* e_in   = (const float*)d_in[1];
  const float* g_n1   = (const float*)d_in[2];
  const float* b_n1   = (const float*)d_in[3];
  const float* g_e1   = (const float*)d_in[4];
  const float* b_e1   = (const float*)d_in[5];
  const float* w_qkv  = (const float*)d_in[6];
  const float* w_e    = (const float*)d_in[7];
  const float* w_g    = (const float*)d_in[8];
  const float* o_n    = (const float*)d_in[9];
  const float* o_e    = (const float*)d_in[10];
  const float* g_n2   = (const float*)d_in[11];
  const float* b_n2   = (const float*)d_in[12];
  const float* w1_n   = (const float*)d_in[13];
  const float* w2_n   = (const float*)d_in[14];
  const float* g_e2   = (const float*)d_in[15];
  const float* b_e2   = (const float*)d_in[16];
  const float* w1_e   = (const float*)d_in[17];
  const float* w2_e   = (const float*)d_in[18];

  float* outN = (float*)d_out;
  float* outE = outN + (size_t)BB * NS * DD;

  float* Ews = (float*)d_ws;                                  // B*H*N*N
  float* Qw  = Ews + (size_t)BB * HH * NS * NS;
  float* Kw  = Qw + (size_t)BB * HH * NS * DKK;
  float* Vw  = Kw + (size_t)BB * HH * NS * DKK;
  float* dyn = Vw + (size_t)BB * HH * NS * DKK;
  float* Vc  = dyn + (size_t)BB * HH * NS;

  k_node_qkv<<<BB * NS, 64, 0, stream>>>(n_in_p, g_n1, b_n1, w_qkv, Qw, Kw, Vw);
  k_edge_proj<<<BB * NS, 256, 0, stream>>>(e_in, g_e1, b_e1, w_e, w_g, Ews, dyn);
  k_attn<<<BB * HH * NS, 256, 0, stream>>>(Qw, Kw, Vw, Ews, dyn, Vc);
  k_node_out<<<BB * NS, 64, 0, stream>>>(n_in_p, Vc, o_n, g_n2, b_n2, w1_n, w2_n, outN);
  k_edge_out<<<BB * NS * (NS / 64), EWAVES * 32, 0, stream>>>(
      e_in, Ews, o_e, g_e2, b_e2, w1_e, w2_e, outE);
}
